// Model_GNN_29454885716684
// MI455X (gfx1250) — compile-verified, matrix operand order checked by reading, not run on hardware
//
#include <hip/hip_runtime.h>
#include <math.h>

// ---------------------------------------------------------------------------
// GatedGraphConv (2 layers) for MI455X / gfx1250, bf16 WMMA with f32 accum.
//   x [9728, 2116] fp32 -> per layer:
//     xs   = neighbor pre-aggregate(x)            (scatter commutes with GEMM)
//     agg  = xs @ W[l]                            (WMMA GEMM, 32x32 per wave)
//     gates: c_r = agg@Wih_r + x@Whh_r, c_z likewise (shared accumulators),
//            c_n = agg@Wih_n, c_hn = x@Whh_n      (fused WMMA GEMM)
//     x'   = GRU epilogue                          (sigmoid/tanh in-register)
// All weights stored N-major [n][k] so BOTH operand tiles live in LDS as
// [row][K]: async b128 fills (GLOBAL_LOAD_ASYNC_TO_LDS_B128, ASYNCcnt) and
// fragment gathers that merge into ds_load_b128 (no u16/mov packing).
// ---------------------------------------------------------------------------

#define BATCH   512
#define NNODES  19
#define ROWS    (BATCH * NNODES)   // 9728 = 76 * 128
#define FEAT    2116
#define FP      2176               // padded: 68*32, 34*64, 136*16
#define KSTEPS  (FP / 32)          // 68
#define NEDGES  36

typedef __attribute__((ext_vector_type(16))) __bf16 v16bf;
typedef __attribute__((ext_vector_type(8)))  __bf16 v8bf;
typedef __attribute__((ext_vector_type(8)))  float  v8f;
typedef int av4i __attribute__((vector_size(4 * sizeof(int))));

__device__ __forceinline__ __bf16 bfc(float f) { return (__bf16)f; }

#if defined(__has_builtin)
#if __has_builtin(__builtin_amdgcn_global_load_async_to_lds_b128) && \
    __has_builtin(__builtin_amdgcn_s_wait_asynccnt)
#define USE_ASYNC_LDS 1
#endif
#endif

// copy 8 bf16 (16 bytes) global -> LDS
__device__ __forceinline__ void cp_lds16(__bf16* dst, const __bf16* src) {
#ifdef USE_ASYNC_LDS
  __builtin_amdgcn_global_load_async_to_lds_b128((av4i*)src, (av4i*)dst, 0, 0);
#else
  *(v8bf*)dst = *(const v8bf*)src;
#endif
}
__device__ __forceinline__ void lds_fill_fence() {
#ifdef USE_ASYNC_LDS
  __builtin_amdgcn_s_wait_asynccnt(0);
#endif
}

// K index inside the 16x32 (A) / 32x16 (B) bf16 operand fragment,
// per CDNA5 ISA 7.12.2 (lane half -> K mapping).
__device__ __forceinline__ int frag_k(int lane, int e) {
  int v = e >> 1, p = e & 1;
  return ((lane >> 4) << 3) + ((v >> 2) << 4) + ((v & 3) << 1) + p;
}

// Operand fragment from an LDS tile T[row][32]:
//  A operand: row = M (rows striped over lanes, K over halves)
//  B operand: row = N (same striping; weights are stored N-major)
// Halves walk K in contiguous 8-half groups -> merges into 2x ds_load_b128.
__device__ __forceinline__ v16bf loadFrag(const __bf16 (*T)[32], int row0, int lane) {
  v16bf f;
  int m = row0 + (lane & 15);
  #pragma unroll
  for (int e = 0; e < 16; ++e) f[e] = T[m][frag_k(lane, e)];
  return f;
}

#define WMMA_BF16(A, B, C) \
  __builtin_amdgcn_wmma_f32_16x16x32_bf16(false, A, false, B, (short)0, C, false, false)

// ------------------------- one-time prep kernels ---------------------------

// weight [2, FEAT, FEAT] (k-major) fp32 -> Wbf [2, FP(n), FP(k)] bf16, 0-pad:
// Wbf[l][n][k] = weight[l][k][n]
__global__ __launch_bounds__(256) void prep_w_kernel(const float* __restrict__ w,
                                                     __bf16* __restrict__ out) {
  int k  = blockIdx.y * blockDim.x + threadIdx.x;
  int ln = blockIdx.x;                 // l*FP + n
  if (k >= FP) return;
  int l = ln / FP, n = ln % FP;
  float v = (k < FEAT && n < FEAT) ? w[((size_t)l * FEAT + k) * FEAT + n] : 0.f;
  out[(size_t)ln * FP + k] = bfc(v);
}

// w [3*FEAT, FEAT] fp32 -> out[g][n][k] = w[g*FEAT + n][k]  (coalesced copy)
__global__ __launch_bounds__(256) void prep_wg_kernel(const float* __restrict__ w,
                                                      __bf16* __restrict__ out) {
  int k  = blockIdx.y * blockDim.x + threadIdx.x;
  int gn = blockIdx.x;                 // g*FP + n
  if (k >= FP) return;
  int g = gn / FP, n = gn % FP;
  float v = (k < FEAT && n < FEAT) ? w[((size_t)g * FEAT + n) * FEAT + k] : 0.f;
  out[(size_t)gn * FP + k] = bfc(v);
}

// x [ROWS, FEAT] fp32 -> xb [ROWS, FP] bf16 (0-pad)
__global__ __launch_bounds__(256) void seed_xbf_kernel(const float* __restrict__ x,
                                                       __bf16* __restrict__ xb) {
  int c = blockIdx.y * blockDim.x + threadIdx.x;
  int r = blockIdx.x;
  if (c >= FP) return;
  xb[(size_t)r * FP + c] = bfc(c < FEAT ? x[(size_t)r * FEAT + c] : 0.f);
}

// xs[b,node,:] = sum over edges(dst==node) x[b,src,:]   (bf16, padded)
__global__ __launch_bounds__(256) void aggregate_kernel(const float* __restrict__ x,
                                                        const int* __restrict__ ei,
                                                        __bf16* __restrict__ xs) {
  __shared__ int ssrc[NEDGES], sdst[NEDGES];
  if (threadIdx.x < NEDGES) {
    ssrc[threadIdx.x] = ei[threadIdx.x];
    sdst[threadIdx.x] = ei[NEDGES + threadIdx.x];
  }
  __syncthreads();
  int c = blockIdx.y * blockDim.x + threadIdx.x;
  int row = blockIdx.x;                 // b*NNODES + node
  if (c >= FP) return;
  float s = 0.f;
  if (c < FEAT) {
    int b = row / NNODES, node = row % NNODES;
    #pragma unroll 1
    for (int e = 0; e < NEDGES; ++e)
      if (sdst[e] == node)
        s += x[(size_t)(b * NNODES + ssrc[e]) * FEAT + c];
  }
  xs[(size_t)row * FP + c] = bfc(s);
}

// --------------------------- WMMA GEMM: agg = xs @ W -----------------------
// 256 threads = 8 wave32 (4 row-groups x 2 col-groups). WG tile 128x64,
// wave tile 32x32 (4 WMMA per K-step, A/B fragments each reused twice).

__global__ __launch_bounds__(256)
void gemm_agg_kernel(const __bf16* __restrict__ A,   // [ROWS][FP]
                     const __bf16* __restrict__ W,   // [FP(n)][FP(k)]
                     __bf16* __restrict__ C) {       // [ROWS][FP]
  __shared__ alignas(16) __bf16 As[2][128][32];      // [m][k]
  __shared__ alignas(16) __bf16 Bs[2][64][32];       // [n][k]

  const int tid  = threadIdx.x;
  const int lane = tid & 31;
  const int wave = tid >> 5;
  const int rowBase = blockIdx.y * 128;
  const int colBase = blockIdx.x * 64;
  const int wRow = (wave >> 1) * 32;    // 0,32,64,96
  const int wCol = (wave & 1) * 32;     // 0,32

  v8f a00 = {}, a01 = {}, a10 = {}, a11 = {};

  auto fillA = [&](int buf, int k0) {
    int r  = tid >> 1;                  // 0..127
    int ko = (tid & 1) * 16;            // 0,16
    const __bf16* src = A + (size_t)(rowBase + r) * FP + k0 + ko;
    cp_lds16(&As[buf][r][ko], src);
    cp_lds16(&As[buf][r][ko + 8], src + 8);
  };
  auto fillB = [&](int buf, int k0) {
    int n  = tid >> 2;                  // 0..63
    int ko = (tid & 3) * 8;             // 0,8,16,24
    cp_lds16(&Bs[buf][n][ko], W + (size_t)(colBase + n) * FP + k0 + ko);
  };

  fillA(0, 0);
  fillB(0, 0);
  lds_fill_fence();
  __syncthreads();

  for (int s = 0; s < KSTEPS; ++s) {
    int cur = s & 1;
    if (s + 1 < KSTEPS) { fillA(cur ^ 1, (s + 1) * 32); fillB(cur ^ 1, (s + 1) * 32); }

    v16bf fa0 = loadFrag(As[cur], wRow, lane);
    v16bf fa1 = loadFrag(As[cur], wRow + 16, lane);
    v16bf fb0 = loadFrag(Bs[cur], wCol, lane);
    v16bf fb1 = loadFrag(Bs[cur], wCol + 16, lane);
    a00 = WMMA_BF16(fa0, fb0, a00);
    a01 = WMMA_BF16(fa0, fb1, a01);
    a10 = WMMA_BF16(fa1, fb0, a10);
    a11 = WMMA_BF16(fa1, fb1, a11);

    lds_fill_fence();
    __syncthreads();
  }

  int n  = lane & 15;
  int mb = (lane >> 4) << 3;
  auto store = [&](const v8f& acc, int ri, int ci) {
    #pragma unroll
    for (int j = 0; j < 8; ++j) {
      int gr = rowBase + wRow + ri + mb + j;
      int gc = colBase + wCol + ci + n;
      C[(size_t)gr * FP + gc] = bfc(acc[j]);
    }
  };
  store(a00, 0, 0); store(a01, 0, 16); store(a10, 16, 0); store(a11, 16, 16);
}

// -------- fused gate GEMMs (4 logical gates) + GRU epilogue ----------------
// WG tile 128 rows x 32 cols, wave tile 32x16: 12 WMMA per K-step against
// 4 A-fragment + 6 B-fragment loads. r/z gates accumulate gi+gh in one C.

__global__ __launch_bounds__(256)
void gemm_gru_kernel(const __bf16* __restrict__ Aagg,  // [ROWS][FP]
                     const __bf16* __restrict__ Ahid,  // [ROWS][FP]
                     const __bf16* __restrict__ Wih,   // [3][FP(n)][FP(k)]
                     const __bf16* __restrict__ Whh,   // [3][FP(n)][FP(k)]
                     const float* __restrict__ b_ih,   // [3*FEAT]
                     const float* __restrict__ b_hh,   // [3*FEAT]
                     const float* __restrict__ xcur,   // [ROWS][FEAT]
                     float* __restrict__ xnext,        // [ROWS][FEAT]
                     __bf16* __restrict__ xbf_out) {   // [ROWS][FP]
  __shared__ alignas(16) __bf16 Aa[2][128][32];        // [m][k]
  __shared__ alignas(16) __bf16 Ax[2][128][32];        // [m][k]
  __shared__ alignas(16) __bf16 Bt[2][6][32][32];      // [g][n][k]

  const int tid  = threadIdx.x;
  const int lane = tid & 31;
  const int wave = tid >> 5;
  const int rowBase = blockIdx.y * 128;
  const int colBase = blockIdx.x * 32;
  const int wRow = (wave >> 1) * 32;    // 0,32,64,96
  const int wCol = (wave & 1) * 16;     // 0,16

  v8f c0_r = {}, c0_z = {}, c0_n = {}, c0_hn = {};
  v8f c1_r = {}, c1_z = {}, c1_n = {}, c1_hn = {};

  auto fillA = [&](int buf, int k0) {
    int r  = tid >> 1;
    int ko = (tid & 1) * 16;
    const __bf16* sa = Aagg + (size_t)(rowBase + r) * FP + k0 + ko;
    const __bf16* sh = Ahid + (size_t)(rowBase + r) * FP + k0 + ko;
    cp_lds16(&Aa[buf][r][ko], sa);
    cp_lds16(&Aa[buf][r][ko + 8], sa + 8);
    cp_lds16(&Ax[buf][r][ko], sh);
    cp_lds16(&Ax[buf][r][ko + 8], sh + 8);
  };
  auto fillB = [&](int buf, int k0) {
    #pragma unroll
    for (int i = 0; i < 3; ++i) {
      int u = tid + i * 256;            // 0..767 units of 8 halfs
      int g = u >> 7;                   // 0..5
      int r = u & 127;
      int n = r >> 2;                   // 0..31
      int ko = (r & 3) * 8;             // 0,8,16,24
      const __bf16* base = (g < 3) ? (Wih + (size_t)g * FP * FP)
                                   : (Whh + (size_t)(g - 3) * FP * FP);
      cp_lds16(&Bt[buf][g][n][ko], base + (size_t)(colBase + n) * FP + k0 + ko);
    }
  };

  fillA(0, 0);
  fillB(0, 0);
  lds_fill_fence();
  __syncthreads();

  for (int s = 0; s < KSTEPS; ++s) {
    int cur = s & 1;
    if (s + 1 < KSTEPS) { fillA(cur ^ 1, (s + 1) * 32); fillB(cur ^ 1, (s + 1) * 32); }

    v16bf af0 = loadFrag(Aa[cur], wRow, lane);
    v16bf af1 = loadFrag(Aa[cur], wRow + 16, lane);
    v16bf hf0 = loadFrag(Ax[cur], wRow, lane);
    v16bf hf1 = loadFrag(Ax[cur], wRow + 16, lane);

    { v16bf b = loadFrag(Bt[cur][0], wCol, lane);
      c0_r  = WMMA_BF16(af0, b, c0_r);  c1_r  = WMMA_BF16(af1, b, c1_r); }
    { v16bf b = loadFrag(Bt[cur][1], wCol, lane);
      c0_z  = WMMA_BF16(af0, b, c0_z);  c1_z  = WMMA_BF16(af1, b, c1_z); }
    { v16bf b = loadFrag(Bt[cur][2], wCol, lane);
      c0_n  = WMMA_BF16(af0, b, c0_n);  c1_n  = WMMA_BF16(af1, b, c1_n); }
    { v16bf b = loadFrag(Bt[cur][3], wCol, lane);
      c0_r  = WMMA_BF16(hf0, b, c0_r);  c1_r  = WMMA_BF16(hf1, b, c1_r); }
    { v16bf b = loadFrag(Bt[cur][4], wCol, lane);
      c0_z  = WMMA_BF16(hf0, b, c0_z);  c1_z  = WMMA_BF16(hf1, b, c1_z); }
    { v16bf b = loadFrag(Bt[cur][5], wCol, lane);
      c0_hn = WMMA_BF16(hf0, b, c0_hn); c1_hn = WMMA_BF16(hf1, b, c1_hn); }

    lds_fill_fence();
    __syncthreads();
  }

  int n  = lane & 15;
  int mb = (lane >> 4) << 3;
  auto epilogue = [&](const v8f& cr, const v8f& cz, const v8f& cn,
                      const v8f& chn, int ri) {
    #pragma unroll
    for (int j = 0; j < 8; ++j) {
      int gr = rowBase + wRow + ri + mb + j;
      int gc = colBase + wCol + n;
      if (gc < FEAT) {
        float r  = 1.f / (1.f + expf(-(cr[j] + b_ih[gc] + b_hh[gc])));
        float z  = 1.f / (1.f + expf(-(cz[j] + b_ih[FEAT + gc] + b_hh[FEAT + gc])));
        float hn = chn[j] + b_hh[2 * FEAT + gc];
        float nn = tanhf(cn[j] + b_ih[2 * FEAT + gc] + r * hn);
        float h  = xcur[(size_t)gr * FEAT + gc];
        float hnew = (1.f - z) * nn + z * h;
        xnext[(size_t)gr * FEAT + gc] = hnew;
        xbf_out[(size_t)gr * FP + gc] = bfc(hnew);
      } else {
        xbf_out[(size_t)gr * FP + gc] = bfc(0.f);
      }
    }
  };
  epilogue(c0_r, c0_z, c0_n, c0_hn, 0);
  epilogue(c1_r, c1_z, c1_n, c1_hn, 16);
}

// ------------------------------- launcher ----------------------------------

extern "C" void kernel_launch(void* const* d_in, const int* in_sizes, int n_in,
                              void* d_out, int out_size, void* d_ws, size_t ws_size,
                              hipStream_t stream) {
  (void)in_sizes; (void)n_in; (void)out_size; (void)ws_size;
  const float* cnn    = (const float*)d_in[0];
  const float* weight = (const float*)d_in[1];
  const float* w_ih   = (const float*)d_in[2];
  const float* w_hh   = (const float*)d_in[3];
  const float* b_ih   = (const float*)d_in[4];
  const float* b_hh   = (const float*)d_in[5];
  const int*   ei     = (const int*)d_in[6];   // int32 (JAX x64 off)
  float* out = (float*)d_out;

  char* ws = (char*)d_ws;
  size_t off = 0;
  auto take = [&](size_t bytes) {
    char* p = ws + off;
    off = (off + bytes + 255) & ~(size_t)255;
    return p;
  };
  __bf16* bufA = (__bf16*)take((size_t)ROWS * FP * 2);   // x / xs ping
  __bf16* bufB = (__bf16*)take((size_t)ROWS * FP * 2);   // xs / x pong
  __bf16* bufC = (__bf16*)take((size_t)ROWS * FP * 2);   // agg
  __bf16* Wbf  = (__bf16*)take((size_t)2 * FP * FP * 2); // [l][n][k]
  __bf16* Wih  = (__bf16*)take((size_t)3 * FP * FP * 2); // [g][n][k]
  __bf16* Whh  = (__bf16*)take((size_t)3 * FP * FP * 2); // [g][n][k]
  float*  xping = (float*)take((size_t)ROWS * FEAT * 4);

  dim3 blk(256);
  dim3 gElem(ROWS, (FP + 255) / 256);
  dim3 gW(2 * FP, (FP + 255) / 256);
  dim3 gWT(3 * FP, (FP + 255) / 256);
  dim3 gAgg(FP / 64, ROWS / 128);   // 34 x 76
  dim3 gGru(FP / 32, ROWS / 128);   // 68 x 76

  prep_w_kernel<<<gW,  blk, 0, stream>>>(weight, Wbf);
  prep_wg_kernel<<<gWT, blk, 0, stream>>>(w_ih, Wih);
  prep_wg_kernel<<<gWT, blk, 0, stream>>>(w_hh, Whh);
  seed_xbf_kernel<<<gElem, blk, 0, stream>>>(cnn, bufA);

  // ---- layer 0 ----
  aggregate_kernel<<<gElem, blk, 0, stream>>>(cnn, ei, bufB);
  gemm_agg_kernel<<<gAgg, blk, 0, stream>>>(bufB, Wbf, bufC);
  gemm_gru_kernel<<<gGru, blk, 0, stream>>>(bufC, bufA, Wih, Whh, b_ih, b_hh,
                                            cnn, xping, bufB);
  // ---- layer 1 ----
  aggregate_kernel<<<gElem, blk, 0, stream>>>(xping, ei, bufA);
  gemm_agg_kernel<<<gAgg, blk, 0, stream>>>(bufA, Wbf + (size_t)FP * FP, bufC);
  gemm_gru_kernel<<<gGru, blk, 0, stream>>>(bufC, bufB, Wih, Whh, b_ih, b_hh,
                                            xping, out, bufA);
}
